// ActivationRealQuantLinear_59347858096264
// MI455X (gfx1250) — 8-bit matrix operand order repaired and checked by reading, compiled, not run on hardware
//
#include <hip/hip_runtime.h>
#include <hip/hip_bf16.h>
#include <stdint.h>

typedef int   v8i __attribute__((ext_vector_type(8)));
typedef float v8f __attribute__((ext_vector_type(8)));
typedef int   v4i_vs __attribute__((vector_size(16)));
typedef int   v2i_vs __attribute__((vector_size(8)));

#define TOKENS 2048
#define KDIM   4096
#define OUTN   4096
#define NGROUP 32
#define GSIZE  128

#if defined(__has_builtin)
#if __has_builtin(__builtin_amdgcn_global_load_async_to_lds_b128)
#define HAVE_ASYNC_LDS 1
#endif
#endif

#ifdef HAVE_ASYNC_LDS
typedef __attribute__((address_space(1))) v4i_vs* glob_v4i_ptr;
typedef __attribute__((address_space(3))) v4i_vs* lds_v4i_ptr;
#define ASYNC_COPY_B128(dst_lds, src_glob)                                     \
    __builtin_amdgcn_global_load_async_to_lds_b128(                            \
        (glob_v4i_ptr)(src_glob), (lds_v4i_ptr)(dst_lds), 0, 0)
#if __has_builtin(__builtin_amdgcn_global_load_async_to_lds_b64)
typedef __attribute__((address_space(1))) v2i_vs* glob_v2i_ptr;
typedef __attribute__((address_space(3))) v2i_vs* lds_v2i_ptr;
#define ASYNC_COPY_B64(dst_lds, src_glob)                                      \
    __builtin_amdgcn_global_load_async_to_lds_b64(                             \
        (glob_v2i_ptr)(src_glob), (lds_v2i_ptr)(dst_lds), 0, 0)
#else
#define ASYNC_COPY_B64(dst_lds, src_glob)                                      \
    (*(uint2*)(dst_lds) = *(const uint2*)(src_glob))
#endif
#if __has_builtin(__builtin_amdgcn_s_wait_asynccnt)
#define WAIT_ASYNC() __builtin_amdgcn_s_wait_asynccnt(0)
#else
#define WAIT_ASYNC() asm volatile("s_wait_asynccnt 0" ::: "memory")
#endif
#else
#define ASYNC_COPY_B128(dst_lds, src_glob)                                     \
    (*(uint4*)(dst_lds) = *(const uint4*)(src_glob))
#define ASYNC_COPY_B64(dst_lds, src_glob)                                      \
    (*(uint2*)(dst_lds) = *(const uint2*)(src_glob))
#define WAIT_ASYNC() ((void)0)
#endif

__global__ __launch_bounds__(256) void quantize_kernel(
    const float* __restrict__ x, unsigned char* __restrict__ q,
    float* __restrict__ xscale, float* __restrict__ xzp)
{
    __shared__ float smin[256];
    __shared__ float smax[256];
    const int token = blockIdx.x;
    const int t = threadIdx.x;
    const float* xrow = x + (size_t)token * KDIM;

    float4 v[4];
    float mn = 1e30f, mx = -1e30f;
#pragma unroll
    for (int i = 0; i < 4; ++i) {
        v[i] = ((const float4*)xrow)[t + i * 256];
        mn = fminf(mn, fminf(fminf(v[i].x, v[i].y), fminf(v[i].z, v[i].w)));
        mx = fmaxf(mx, fmaxf(fmaxf(v[i].x, v[i].y), fmaxf(v[i].z, v[i].w)));
    }
    smin[t] = mn; smax[t] = mx;
    __syncthreads();
    for (int s = 128; s > 0; s >>= 1) {
        if (t < s) {
            smin[t] = fminf(smin[t], smin[t + s]);
            smax[t] = fmaxf(smax[t], smax[t + s]);
        }
        __syncthreads();
    }
    mn = smin[0]; mx = smax[0];

    const float scale = fmaxf((mx - mn) * (1.0f / 255.0f), 1e-5f);
    const float zp = fminf(fmaxf(rintf(-mn / scale), 0.0f), 255.0f);

    unsigned char* qrow = q + (size_t)token * KDIM;
#pragma unroll
    for (int i = 0; i < 4; ++i) {
        uchar4 o;
        o.x = (unsigned char)fminf(fmaxf(rintf(v[i].x / scale) + zp, 0.0f), 255.0f);
        o.y = (unsigned char)fminf(fmaxf(rintf(v[i].y / scale) + zp, 0.0f), 255.0f);
        o.z = (unsigned char)fminf(fmaxf(rintf(v[i].z / scale) + zp, 0.0f), 255.0f);
        o.w = (unsigned char)fminf(fmaxf(rintf(v[i].w / scale) + zp, 0.0f), 255.0f);
        ((uchar4*)qrow)[t + i * 256] = o;
    }
    if (t == 0) {
        xscale[token] = scale;
        xzp[token] = zp;
    }
}

__global__ __launch_bounds__(256) void packw_kernel(
    const int* __restrict__ qw, const float* __restrict__ wsc,
    const int* __restrict__ wzp, signed char* __restrict__ pw,
    float* __restrict__ wcorr)
{
    const int wid = threadIdx.x >> 5;
    const int lane = threadIdx.x & 31;
    const int o = blockIdx.x * 8 + wid;
    const int g = lane;

    const int zp = wzp[o * NGROUP + g];
    const int* src = qw + ((size_t)o * NGROUP + g) * GSIZE;
    unsigned int* dst = (unsigned int*)(pw + (size_t)o * KDIM + g * GSIZE);

    int csum = 0;
#pragma unroll 4
    for (int c = 0; c < GSIZE; c += 4) {
        int4 cd = ((const int4*)src)[c >> 2];
        csum += cd.x + cd.y + cd.z + cd.w;
        unsigned int packed =  ((unsigned)((cd.x - zp) & 0xFF))
                            | (((unsigned)((cd.y - zp) & 0xFF)) << 8)
                            | (((unsigned)((cd.z - zp) & 0xFF)) << 16)
                            | (((unsigned)((cd.w - zp) & 0xFF)) << 24);
        dst[c >> 2] = packed;
    }
    const int wsum = csum - GSIZE * zp;
    float part = wsc[o * NGROUP + g] * (float)wsum;
#pragma unroll
    for (int m = 16; m >= 1; m >>= 1) part += __shfl_xor(part, m, 32);
    if (lane == 0) wcorr[o] = part;
}

__global__ __launch_bounds__(512) void gemm_kernel(
    const unsigned char* __restrict__ q, const signed char* __restrict__ pw,
    const float* __restrict__ wsc, const float* __restrict__ wcorr,
    const float* __restrict__ bias, const float* __restrict__ xscale,
    const float* __restrict__ xzp, float* __restrict__ out)
{
    __shared__ unsigned char As[2][128][80];
    __shared__ unsigned char Bs[2][64][80];
    __shared__ float sWsc[64][33];

    const int tid  = threadIdx.x;
    const int wid  = tid >> 5;
    const int lane = tid & 31;
    const int half = lane >> 4;
    const int l16  = lane & 15;
    const int rb   = wid >> 1;
    const int cb   = (wid & 1) * 32;

    const int nbase = blockIdx.x * 64;
    const int mbase = blockIdx.y * 128;

    const v8i vzero = {};
    v8f facc[2] = {};
    v8i accA[2] = {};
    v8i accB[2] = {};

#pragma unroll
    for (int i = tid; i < 64 * 32; i += 512)
        sWsc[i >> 5][i & 31] = wsc[(size_t)nbase * NGROUP + i];

    const int arow = tid >> 2, aoff = (tid & 3) * 16;
    const int bcol2 = tid >> 3, boff = (tid & 7) * 8;
    const unsigned char* gA = q  + (size_t)(mbase + arow)  * KDIM + aoff;
    const unsigned char* gB = (const unsigned char*)pw
                            + (size_t)(nbase + bcol2) * KDIM + boff;

    auto stage = [&](int buf) {
        ASYNC_COPY_B128(&As[buf][arow][aoff], gA);
        ASYNC_COPY_B64(&Bs[buf][bcol2][boff], gB);
        gA += 64; gB += 64;
    };

    auto kstep = [&](int buf, v8i* acc, bool first) {
        stage(buf ^ 1);
        v8i a;
        {
            const unsigned char* arowp = &As[buf][rb * 16 + l16][0];
            unsigned long long a01 = *(const unsigned long long*)(arowp +      16 * half);
            unsigned long long a23 = *(const unsigned long long*)(arowp + 8  + 16 * half);
            unsigned long long a45 = *(const unsigned long long*)(arowp + 32 + 16 * half);
            unsigned long long a67 = *(const unsigned long long*)(arowp + 40 + 16 * half);
            a[0] = (int)(unsigned)a01; a[1] = (int)(unsigned)(a01 >> 32);
            a[2] = (int)(unsigned)a23; a[3] = (int)(unsigned)(a23 >> 32);
            a[4] = (int)(unsigned)a45; a[5] = (int)(unsigned)(a45 >> 32);
            a[6] = (int)(unsigned)a67; a[7] = (int)(unsigned)(a67 >> 32);
        }
#pragma unroll
        for (int t = 0; t < 2; ++t) {
            const unsigned char* bcolp = &Bs[buf][cb + t * 16 + l16][0];
            uint4 b03 = *(const uint4*)(bcolp +      16 * half);
            uint4 b47 = *(const uint4*)(bcolp + 32 + 16 * half);
            v8i b;
            b[0] = (int)b03.x; b[1] = (int)b03.y; b[2] = (int)b03.z; b[3] = (int)b03.w;
            b[4] = (int)b47.x; b[5] = (int)b47.y; b[6] = (int)b47.z; b[7] = (int)b47.w;
            acc[t] = __builtin_amdgcn_wmma_i32_16x16x64_iu8(
                false, a, true, b, first ? vzero : acc[t], false, false);
        }
        WAIT_ASYNC();
        __syncthreads();
    };

    auto flush = [&](const v8i* acc, int gidx) {
#pragma unroll
        for (int t = 0; t < 2; ++t) {
            const float ws = sWsc[cb + t * 16 + l16][gidx];
#pragma unroll
            for (int r = 0; r < 8; ++r)
                facc[t][r] += ws * (float)acc[t][r];
        }
    };

    stage(0);
    WAIT_ASYNC();
    __syncthreads();

    for (int kg2 = 0; kg2 < NGROUP / 2; ++kg2) {
        kstep(0, accA, true);
        kstep(1, accA, false);
        flush(accB, kg2 == 0 ? 0 : 2 * kg2 - 1);
        kstep(0, accB, true);
        kstep(1, accB, false);
        flush(accA, 2 * kg2);
    }
    flush(accB, NGROUP - 1);

#pragma unroll
    for (int t = 0; t < 2; ++t) {
        const int n = nbase + cb + t * 16 + l16;
        const float wc = wcorr[n];
        const float bn = bias[n];
#pragma unroll
        for (int r = 0; r < 8; ++r) {
            const int m = mbase + rb * 16 + r + 8 * half;
            const float val = xscale[m] * (facc[t][r] - xzp[m] * wc) + bn;
            out[(size_t)m * OUTN + n] = val;
        }
    }
}

extern "C" void kernel_launch(void* const* d_in, const int* in_sizes, int n_in,
                              void* d_out, int out_size, void* d_ws, size_t ws_size,
                              hipStream_t stream) {
    const float* x    = (const float*)d_in[0];
    const int*   qw   = (const int*)d_in[1];
    const float* wsc  = (const float*)d_in[2];
    const int*   wzp  = (const int*)d_in[3];
    const float* bias = (const float*)d_in[4];
    float* out = (float*)d_out;

    char* ws = (char*)d_ws;
    unsigned char* q  = (unsigned char*)ws;
    signed char*   pw = (signed char*)(ws + (8u << 20));
    float* xs    = (float*)(ws + (24u << 20));
    float* xzp   = xs + TOKENS;
    float* wcorr = xzp + TOKENS;

    quantize_kernel<<<TOKENS, 256, 0, stream>>>(x, q, xs, xzp);
    packw_kernel<<<OUTN / 8, 256, 0, stream>>>(qw, wsc, wzp, pw, wcorr);
    dim3 grid(OUTN / 64, TOKENS / 128);
    gemm_kernel<<<grid, 512, 0, stream>>>(q, pw, wsc, wcorr, bias, xs, xzp, out);
}
